// STHGATLikeModel_64656437674250
// MI455X (gfx1250) — compile-verified, matrix operand order checked by reading, not run on hardware
//
#include <hip/hip_runtime.h>
#include <math.h>

// ---------------------------------------------------------------------------
// Problem constants (from the reference)
// ---------------------------------------------------------------------------
constexpr int NP   = 4096;    // pieces
constexpr int NS   = 8192;    // squares
constexpr int NB   = 128;     // batch
constexpr int IN0  = 10;
constexpr int HD   = 512;     // hidden
constexpr int NH   = 4;       // heads
constexpr int DKK  = 128;     // per-head dim
constexpr int EINT = 98304;   // all edge counts are multiples of 16
constexpr int ERAY = 65536;
constexpr int EPS  = 32768;
constexpr int ESP  = 32768;
constexpr float ATT_SCALE = 0.08838834764831845f; // 1/sqrt(128)

typedef __attribute__((ext_vector_type(16))) _Float16 v16h;
typedef __attribute__((ext_vector_type(8)))  _Float16 v8h;
typedef __attribute__((ext_vector_type(4)))  _Float16 v4h;
typedef __attribute__((ext_vector_type(8)))  float    v8f;

// ---------------------------------------------------------------------------
// A-fragment load from an f16 row (CDNA5 16x16x32 f16 A layout, ISA 7.12.2).
// Lane's 16 K-values = row[kb..kb+7] ++ row[kb+16..kb+23]  (kb = 0 or 8)
// -> exactly 2 global_load_b128, no converts.
// ---------------------------------------------------------------------------
__device__ inline v16h load_a_frag_f16(const _Float16* __restrict__ row) {
  int lane = threadIdx.x & 31;
  int kb = (lane & 16) ? 8 : 0;
  v8h lo = *(const v8h*)(row + kb);
  v8h hi = *(const v8h*)(row + kb + 16);
  return __builtin_shufflevector(lo, hi, 0, 1, 2, 3, 4, 5, 6, 7,
                                 8, 9, 10, 11, 12, 13, 14, 15);
}

__device__ inline void atomicMaxFloat(float* addr, float val) {
  int* ai = (int*)addr;
  int cur = __float_as_int(*addr);
  while (__int_as_float(cur) < val) {
    int old = atomicCAS(ai, cur, __float_as_int(val));
    if (old == cur) break;
    cur = old;
  }
}

__device__ inline float sigmoidf_(float x) { return 1.0f / (1.0f + expf(-x)); }

// ---------------------------------------------------------------------------
// Elementwise f32 -> f16 convert (vectorized: one float4 per thread).
// ---------------------------------------------------------------------------
__global__ void cvt_f16_kernel(const float* __restrict__ s,
                               _Float16* __restrict__ d, size_t n) {
  size_t i4 = ((size_t)blockIdx.x * blockDim.x + threadIdx.x) * 4;
  if (i4 >= n) return;
  float4 v = *(const float4*)(s + i4);
  v4h o;
  o[0] = (_Float16)v.x; o[1] = (_Float16)v.y;
  o[2] = (_Float16)v.z; o[3] = (_Float16)v.w;
  *(v4h*)(d + i4) = o;
}

// ---------------------------------------------------------------------------
// Pre-pack a KxN f32 matrix (or NxK if transB) into f16 WMMA-B fragments.
// Fragment (nt,kt) at out[((nt*KT + kt)*32 + lane)*16]; consumer loads one
// aligned v16h per fragment. grid.y batches matrices (attention heads).
// ---------------------------------------------------------------------------
__global__ void pack_b_kernel(const float* __restrict__ W,
                              _Float16* __restrict__ out,
                              int K, int N, int transB) {
  W   += (size_t)blockIdx.y * K * N;
  out += (size_t)blockIdx.y * K * N;
  int KT = K >> 5;
  int NT = N >> 4;
  int idx = blockIdx.x * (blockDim.x >> 5) + (threadIdx.x >> 5);
  if (idx >= KT * NT) return;
  int nt = idx / KT, kt = idx % KT;
  int lane = threadIdx.x & 31;
  int n = nt * 16 + (lane & 15);
  int kb = (lane & 16) ? 8 : 0;
  _Float16* dst = out + ((size_t)idx * 32 + lane) * 16;
#pragma unroll
  for (int j = 0; j < 8; ++j) {
#pragma unroll
    for (int p = 0; p < 2; ++p) {
      int kk = kb + ((j < 4) ? (2 * j + p) : (16 + 2 * (j - 4) + p));
      int k = kt * 32 + kk;
      float v = transB ? W[(size_t)n * K + k] : W[(size_t)k * N + n];
      dst[2 * j + p] = (_Float16)v;
    }
  }
}

// ---------------------------------------------------------------------------
// WMMA GEMM: C = act(A16[MxK] @ Wp + bias [+ skip]); optional f16 mirror C16.
// 128 threads = 4 waves; each wave computes a 32x64 macro-tile (2 M-tiles x
// 4 N-tiles = 8 WMMAs per K-step from 4 A-loads + 8 B-loads). Block: 32x256.
// K % 32 == 0, M % 32 == 0, N % 256 == 0.
// ---------------------------------------------------------------------------
__global__ void wmma_gemm_kernel(const _Float16* __restrict__ A16,
                                 const _Float16* __restrict__ Wp,
                                 const float* __restrict__ bias,
                                 const float* __restrict__ skip,
                                 float* __restrict__ C,
                                 _Float16* __restrict__ C16,
                                 int M, int K, int N, int act) {
  int wave = threadIdx.x >> 5;
  int lane = threadIdx.x & 31;
  int n0 = blockIdx.x * 256 + wave * 64;
  int m0 = blockIdx.y * 32;
  if (n0 >= N) return;
  const _Float16* Arow0 = A16 + (size_t)(m0 + (lane & 15)) * K;
  const _Float16* Arow1 = Arow0 + (size_t)16 * K;
  int KT = K >> 5;
  const _Float16* wp = Wp + (size_t)lane * 16;
  size_t bidx0 = (size_t)(n0 >> 4) * KT;
  v8f zero = {};
  v8f acc[8];
#pragma unroll
  for (int i = 0; i < 8; ++i) acc[i] = zero;
  for (int kt = 0; kt < KT; ++kt) {
    __builtin_prefetch(Arow0 + kt * 32 + 128, 0, 1);  // global_prefetch_b8
    v16h af0 = load_a_frag_f16(Arow0 + kt * 32);
    v16h af1 = load_a_frag_f16(Arow1 + kt * 32);
#pragma unroll
    for (int t = 0; t < 4; ++t) {
      v16h bf = *(const v16h*)(wp + (bidx0 + (size_t)t * KT + kt) * 512);
      acc[t] = __builtin_amdgcn_wmma_f32_16x16x32_f16(false, af0, false, bf,
                                                      (short)0, acc[t], false, false);
      acc[4 + t] = __builtin_amdgcn_wmma_f32_16x16x32_f16(false, af1, false, bf,
                                                          (short)0, acc[4 + t], false, false);
    }
  }
  int nl = lane & 15;
  int mhalf = (lane & 16) ? 8 : 0;
#pragma unroll
  for (int mi = 0; mi < 2; ++mi) {
    int mb = m0 + mi * 16 + mhalf;
#pragma unroll
    for (int t = 0; t < 4; ++t) {
      int n = n0 + t * 16 + nl;
      float bv = bias[n];
#pragma unroll
      for (int r = 0; r < 8; ++r) {
        int m = mb + r;
        float v = acc[mi * 4 + t][r] + bv;
        if (skip) v += skip[(size_t)m * N + n];
        if (act) v = 0.5f * v * (1.0f + erff(v * 0.70710678118654752f)); // GELU
        C[(size_t)m * N + n] = v;
        if (C16) C16[(size_t)m * N + n] = (_Float16)v;
      }
    }
  }
}

// ---------------------------------------------------------------------------
// Edge attention logits: alpha[e,h] = (k_j @ att_h) . q_i * pri * scale (*(1+ew))
// 16 edges per wave per head. A fragments (f16, gathered) hoisted: loaded
// ONCE and reused across all 8 N-tiles (32 WMMAs). att pre-packed f16.
// ---------------------------------------------------------------------------
__global__ void edge_alpha_kernel(const _Float16* __restrict__ k16,
                                  const float* __restrict__ qdst,
                                  const int* __restrict__ src,
                                  const int* __restrict__ dst,
                                  const _Float16* __restrict__ attpk,
                                  const float* __restrict__ pri,
                                  const float* __restrict__ ew, int ewstride,
                                  float scale,
                                  float* __restrict__ alpha,
                                  float* __restrict__ segmax) {
  int h = blockIdx.y;
  int e0 = blockIdx.x * 16;
  int lane = threadIdx.x & 31;
  const _Float16* abase = k16 + (size_t)src[e0 + (lane & 15)] * HD + h * DKK;
  const _Float16* atth = attpk + (size_t)h * DKK * DKK + (size_t)lane * 16;
  int n_lane = lane & 15;
  int mb = (lane & 16) ? 8 : 0;
  v16h afr[4];
#pragma unroll
  for (int kt = 0; kt < 4; ++kt) afr[kt] = load_a_frag_f16(abase + kt * 32);
  const float* qrow[8];
  int drow[8];
#pragma unroll
  for (int r = 0; r < 8; ++r) {
    drow[r] = dst[e0 + mb + r];
    qrow[r] = qdst + (size_t)drow[r] * HD + h * DKK;
  }
  float part[8];
#pragma unroll
  for (int r = 0; r < 8; ++r) part[r] = 0.0f;
  for (int nt = 0; nt < 8; ++nt) {
    v8f acc = {};
#pragma unroll
    for (int kt = 0; kt < 4; ++kt) {
      v16h bf = *(const v16h*)(atth + (size_t)(nt * 4 + kt) * 512);
      acc = __builtin_amdgcn_wmma_f32_16x16x32_f16(false, afr[kt], false, bf,
                                                   (short)0, acc, false, false);
    }
    int n = nt * 16 + n_lane;
#pragma unroll
    for (int r = 0; r < 8; ++r) part[r] += acc[r] * qrow[r][n];
  }
#pragma unroll
  for (int r = 0; r < 8; ++r)
    for (int off = 8; off; off >>= 1)
      part[r] += __shfl_xor(part[r], off, 32);
  if ((lane & 15) == 0) {
    float p0 = pri[0];
#pragma unroll
    for (int r = 0; r < 8; ++r) {
      int e = e0 + mb + r;
      float a = part[r] * p0 * scale;
      if (ew) a *= (1.0f + ew[(size_t)e * ewstride]);
      alpha[(size_t)e * NH + h] = a;
      atomicMaxFloat(&segmax[(size_t)drow[r] * NH + h], a);
    }
  }
}

__global__ void edge_exp_kernel(float* __restrict__ alpha,
                                const int* __restrict__ dst,
                                const float* __restrict__ segmax,
                                float* __restrict__ segsum, int E4) {
  int i = blockIdx.x * blockDim.x + threadIdx.x;
  if (i >= E4) return;
  int e = i >> 2, h = i & 3;
  int d = dst[e];
  float ea = expf(alpha[i] - segmax[(size_t)d * NH + h]);
  alpha[i] = ea;
  atomicAdd(&segsum[(size_t)d * NH + h], ea);
}

__global__ void edge_norm_kernel(float* __restrict__ alpha,
                                 const int* __restrict__ dst,
                                 const float* __restrict__ segsum, int E4) {
  int i = blockIdx.x * blockDim.x + threadIdx.x;
  if (i >= E4) return;
  int e = i >> 2, h = i & 3;
  alpha[i] = alpha[i] / (segsum[(size_t)dst[e] * NH + h] + 1e-16f);
}

// v_msg = (v_j @ msg_h) * a ; scatter-add into out[dst]; A hoisted like above.
__global__ void edge_vmsg_kernel(const _Float16* __restrict__ v16s,
                                 const int* __restrict__ src,
                                 const int* __restrict__ dst,
                                 const _Float16* __restrict__ msgpk,
                                 const float* __restrict__ a,
                                 float* __restrict__ out) {
  int h = blockIdx.y;
  int e0 = blockIdx.x * 16;
  int lane = threadIdx.x & 31;
  const _Float16* abase = v16s + (size_t)src[e0 + (lane & 15)] * HD + h * DKK;
  const _Float16* msgh = msgpk + (size_t)h * DKK * DKK + (size_t)lane * 16;
  int n_lane = lane & 15;
  int mb = (lane & 16) ? 8 : 0;
  v16h afr[4];
#pragma unroll
  for (int kt = 0; kt < 4; ++kt) afr[kt] = load_a_frag_f16(abase + kt * 32);
  float av[8];
  float* orow[8];
#pragma unroll
  for (int r = 0; r < 8; ++r) {
    int e = e0 + mb + r;
    av[r] = a[(size_t)e * NH + h];
    orow[r] = out + (size_t)dst[e] * HD + h * DKK;
  }
  for (int nt = 0; nt < 8; ++nt) {
    v8f acc = {};
#pragma unroll
    for (int kt = 0; kt < 4; ++kt) {
      v16h bf = *(const v16h*)(msgh + (size_t)(nt * 4 + kt) * 512);
      acc = __builtin_amdgcn_wmma_f32_16x16x32_f16(false, afr[kt], false, bf,
                                                   (short)0, acc, false, false);
    }
    int n = nt * 16 + n_lane;
#pragma unroll
    for (int r = 0; r < 8; ++r)
      atomicAdd(&orow[r][n], acc[r] * av[r]);
  }
}

// ---------------------------------------------------------------------------
// Small supporting kernels
// ---------------------------------------------------------------------------
__global__ void fill_kernel(float* __restrict__ p, float v, size_t n) {
  size_t i = (size_t)blockIdx.x * blockDim.x + threadIdx.x;
  if (i < n) p[i] = v;
}

// Generic small dense linear (layer-0, K=10); optional f16 mirror output.
__global__ void lin_small_kernel(const float* __restrict__ A,
                                 const float* __restrict__ W,
                                 const float* __restrict__ b,
                                 float* __restrict__ C,
                                 _Float16* __restrict__ C16,
                                 int M, int K, int N) {
  int i = blockIdx.x * blockDim.x + threadIdx.x;
  if (i >= M * N) return;
  int m = i / N, n = i % N;
  float s = b[n];
  for (int k = 0; k < K; ++k) s += A[(size_t)m * K + k] * W[(size_t)k * N + n];
  C[i] = s;
  if (C16) C16[i] = (_Float16)s;
}

__global__ void tval_kernel(const float* __restrict__ xp,
                            const float* __restrict__ tw,
                            const float* __restrict__ tb,
                            float* __restrict__ tval) {
  int wave = threadIdx.x >> 5, lane = threadIdx.x & 31;
  int p = blockIdx.x * 4 + wave;
  const float* row = xp + (size_t)p * HD;
  float s = 0.0f;
  for (int k = lane; k < HD; k += 32) s += row[k] * tw[k];
  for (int off = 16; off; off >>= 1) s += __shfl_xor(s, off, 32);
  if (lane == 0) tval[p] = sigmoidf_(s + tb[0]);
}

__global__ void ray_scatter_kernel(const float* __restrict__ xproj,
                                   const float* __restrict__ earay,
                                   const float* __restrict__ tval,
                                   const int* __restrict__ src,
                                   const int* __restrict__ dst,
                                   float* __restrict__ xout, int E) {
  int e = blockIdx.x;
  if (e >= E) return;
  float dist = earay[(size_t)e * 2], blk = earay[(size_t)e * 2 + 1];
  int s = src[e], d = dst[e];
  float w = 1.0f / (1.0f + blk + 0.1f * dist) * tval[s];
  const float* xr = xproj + (size_t)s * HD;
  float* od = xout + (size_t)d * HD;
  for (int c = threadIdx.x; c < HD; c += blockDim.x)
    atomicAdd(&od[c], xr[c] * w);
}

__global__ void pool_kernel(const float* __restrict__ x, float* __restrict__ pool,
                            int per) {
  int b = blockIdx.x, c = threadIdx.x;
  const float* base = x + (size_t)b * per * HD + c;
  float s = 0.0f;
  for (int i = 0; i < per; ++i) s += base[(size_t)i * HD];
  pool[(size_t)b * HD + c] = s / (float)per;
}

__global__ void concat_kernel(const float* __restrict__ pp,
                              const float* __restrict__ sp,
                              float* __restrict__ g) {
  int idx = blockIdx.x * blockDim.x + threadIdx.x;
  if (idx >= NB * 2 * HD) return;
  int t = idx >> 10, c = idx & 1023;
  g[idx] = (c < HD) ? pp[(size_t)t * HD + c] : sp[(size_t)t * HD + (c - HD)];
}

// Sequential GRU scan, single block of HD threads; h kept in LDS.
__global__ void gru_kernel(const float* __restrict__ GI,
                           const float* __restrict__ Whh,
                           const float* __restrict__ bhh,
                           float* __restrict__ gru_out) {
  __shared__ float hsh[HD];
  int j = threadIdx.x;
  hsh[j] = 0.0f;
  __syncthreads();
  for (int t = 0; t < NB; ++t) {
    float g0 = bhh[j], g1 = bhh[j + HD], g2 = bhh[j + 2 * HD];
    for (int k = 0; k < HD; ++k) {
      float hv = hsh[k];
      g0 += Whh[(size_t)j * HD + k] * hv;
      g1 += Whh[(size_t)(j + HD) * HD + k] * hv;
      g2 += Whh[(size_t)(j + 2 * HD) * HD + k] * hv;
    }
    float i_r = GI[(size_t)t * 3 * HD + j];
    float i_z = GI[(size_t)t * 3 * HD + HD + j];
    float i_n = GI[(size_t)t * 3 * HD + 2 * HD + j];
    float r = sigmoidf_(i_r + g0);
    float z = sigmoidf_(i_z + g1);
    float n = tanhf(i_n + r * g2);
    float hnew = (1.0f - z) * n + z * hsh[j];
    __syncthreads();
    hsh[j] = hnew;
    gru_out[(size_t)t * HD + j] = hnew;
    __syncthreads();
  }
}

// Final heads: win 512->512(relu)->3 ; mat/dom 512->32(relu)->1
__global__ void heads_kernel(const float* __restrict__ gru_out,
    const float* w_w0, const float* b_w0, const float* w_w1, const float* b_w1,
    const float* w_m0, const float* b_m0, const float* w_m1, const float* b_m1,
    const float* w_d0, const float* b_d0, const float* w_d1, const float* b_d1,
    float* __restrict__ out) {
  __shared__ float xv[HD];
  __shared__ float hid[HD];
  __shared__ float h32[32];
  int t = blockIdx.x, j = threadIdx.x;
  xv[j] = gru_out[(size_t)t * HD + j];
  __syncthreads();
  float s = b_w0[j];
  for (int k = 0; k < HD; ++k) s += xv[k] * w_w0[(size_t)k * HD + j];
  hid[j] = fmaxf(s, 0.0f);
  __syncthreads();
  if (j < 3) {
    float o = b_w1[j];
    for (int k = 0; k < HD; ++k) o += hid[k] * w_w1[(size_t)k * 3 + j];
    out[t * 3 + j] = o;
  }
  if (j < 32) {
    float o = b_m0[j];
    for (int k = 0; k < HD; ++k) o += xv[k] * w_m0[(size_t)k * 32 + j];
    h32[j] = fmaxf(o, 0.0f);
  }
  __syncthreads();
  if (j == 0) {
    float o = b_m1[0];
    for (int k = 0; k < 32; ++k) o += h32[k] * w_m1[k];
    out[NB * 3 + t] = o;
  }
  __syncthreads();
  if (j < 32) {
    float o = b_d0[j];
    for (int k = 0; k < HD; ++k) o += xv[k] * w_d0[(size_t)k * 32 + j];
    h32[j] = fmaxf(o, 0.0f);
  }
  __syncthreads();
  if (j == 0) {
    float o = b_d1[0];
    for (int k = 0; k < 32; ++k) o += h32[k] * w_d1[k];
    out[NB * 4 + t] = o;
  }
}

// ---------------------------------------------------------------------------
// Host orchestration
// ---------------------------------------------------------------------------
extern "C" void kernel_launch(void* const* d_in, const int* in_sizes, int n_in,
                              void* d_out, int out_size, void* d_ws, size_t ws_size,
                              hipStream_t stream) {
  (void)in_sizes; (void)out_size; (void)ws_size;
  const float* piece_x  = (const float*)d_in[0];
  const float* square_x = (const float*)d_in[1];
  const int*   ei_int   = (const int*)d_in[2];
  const float* ea_int   = (const float*)d_in[3];
  const int*   ei_ray   = (const int*)d_in[4];
  const float* ea_ray   = (const float*)d_in[5];
  const int*   ei_ps    = (const int*)d_in[6];
  const int*   ei_sp    = (const int*)d_in[7];
  // d_in[8]/d_in[9] (batch ids) unused: batches are contiguous by construction.

  // Params assumed flattened as jax pytree leaves (sorted dict keys) after the
  // 10 data inputs (see round-0 notes for the full layout).
  const float* P[110];
  for (int i = 0; i < 110 && (10 + i) < n_in; ++i) P[i] = (const float*)d_in[10 + i];

  char* wsp = (char*)d_ws;
  size_t off = 0;
  auto ALLOC = [&](size_t elems) -> float* {
    float* p = (float*)(wsp + off);
    off += ((elems * sizeof(float) + 255) & ~(size_t)255);
    return p;
  };
  auto ALLOCH = [&](size_t halves) -> _Float16* {
    _Float16* p = (_Float16*)(wsp + off);
    off += ((halves * sizeof(_Float16) + 255) & ~(size_t)255);
    return p;
  };
  float* sq10   = ALLOC((size_t)NS * IN0);
  float* xp     = ALLOC((size_t)NP * HD);
  float* xs     = ALLOC((size_t)NS * HD);
  float* xp_new = ALLOC((size_t)NP * HD);
  float* xs_new = ALLOC((size_t)NS * HD);
  float* kd_p   = ALLOC((size_t)NP * HD);
  float* qd_p   = ALLOC((size_t)NP * HD);
  float* vd_p   = ALLOC((size_t)NP * HD);
  float* kd_s   = ALLOC((size_t)NS * HD);
  float* qd_s   = ALLOC((size_t)NS * HD);
  float* vd_s   = ALLOC((size_t)NS * HD);
  float* out_p  = ALLOC((size_t)NP * HD);
  float* out_s  = ALLOC((size_t)NS * HD);
  float* skip_p = ALLOC((size_t)NP * HD);
  float* skip_s = ALLOC((size_t)NS * HD);
  float* alpha  = ALLOC((size_t)EINT * NH);
  float* segmax = ALLOC((size_t)NS * NH);
  float* segsum = ALLOC((size_t)NS * NH);
  float* tval   = ALLOC((size_t)NP);
  float* xproj  = ALLOC((size_t)NP * HD);
  float* xp2    = ALLOC((size_t)NP * HD);
  float* ppool  = ALLOC((size_t)NB * HD);
  float* spool  = ALLOC((size_t)NB * HD);
  float* gbuf   = ALLOC((size_t)NB * 2 * HD);
  float* GI     = ALLOC((size_t)NB * 3 * HD);
  float* gruout = ALLOC((size_t)NB * HD);
  _Float16* wpack  = ALLOCH((size_t)1024 * 1536);   // largest weight (W_ih)
  _Float16* attpk  = ALLOCH((size_t)NH * DKK * DKK);
  _Float16* msgpk  = ALLOCH((size_t)NH * DKK * DKK);
  _Float16* xp16   = ALLOCH((size_t)NP * HD);       // f16 mirrors of WMMA A inputs
  _Float16* xs16   = ALLOCH((size_t)NS * HD);
  _Float16* kd_p16 = ALLOCH((size_t)NP * HD);
  _Float16* vd_p16 = ALLOCH((size_t)NP * HD);
  _Float16* kd_s16 = ALLOCH((size_t)NS * HD);
  _Float16* vd_s16 = ALLOCH((size_t)NS * HD);
  _Float16* apack  = ALLOCH((size_t)NS * HD);       // scratch f16 A

  auto pack = [&](const float* W, _Float16* dst, int K, int N, int transB, int batch) {
    int waves = (K >> 5) * (N >> 4);
    dim3 g((waves + 3) / 4, batch);
    pack_b_kernel<<<g, 128, 0, stream>>>(W, dst, K, N, transB);
  };
  auto gemmA16 = [&](const _Float16* A16, const float* W, const float* b,
                     const float* skip, float* C, _Float16* C16,
                     int M, int K, int N, int transB, int act) {
    pack(W, wpack, K, N, transB, 1);
    dim3 g((N + 255) / 256, M / 32);
    wmma_gemm_kernel<<<g, 128, 0, stream>>>(A16, wpack, b, skip, C, C16, M, K, N, act);
  };
  auto gemmF32 = [&](const float* A, const float* W, const float* b,
                     const float* skip, float* C, _Float16* C16,
                     int M, int K, int N, int transB, int act) {
    size_t n = (size_t)M * K;
    cvt_f16_kernel<<<(unsigned)((n / 4 + 255) / 256), 256, 0, stream>>>(A, apack, n);
    gemmA16(apack, W, b, skip, C, C16, M, K, N, transB, act);
  };
  auto fill = [&](float* p, float v, size_t n) {
    fill_kernel<<<(unsigned)((n + 255) / 256), 256, 0, stream>>>(p, v, n);
  };
  auto lin10 = [&](const float* A, const float* W, const float* b,
                   float* C, _Float16* C16, int M) {
    lin_small_kernel<<<(M * HD + 255) / 256, 256, 0, stream>>>(A, W, b, C, C16, M, IN0, HD);
  };
  auto process_edges = [&](const int* src, const int* dst, int E,
                           const _Float16* k16, const float* qd, const _Float16* v16s,
                           const float* att, const float* msg, const float* pri,
                           const float* ew, int ewstride, int Ndst, float* ob) {
    pack(att, attpk, DKK, DKK, 0, NH);
    pack(msg, msgpk, DKK, DKK, 0, NH);
    fill(segmax, -1e30f, (size_t)Ndst * NH);
    fill(segsum, 0.0f, (size_t)Ndst * NH);
    dim3 g(E / 16, NH);
    edge_alpha_kernel<<<g, 32, 0, stream>>>(k16, qd, src, dst, attpk, pri, ew,
                                            ewstride, ATT_SCALE, alpha, segmax);
    int E4 = E * NH;
    edge_exp_kernel<<<(E4 + 255) / 256, 256, 0, stream>>>(alpha, dst, segmax, segsum, E4);
    edge_norm_kernel<<<(E4 + 255) / 256, 256, 0, stream>>>(alpha, dst, segsum, E4);
    edge_vmsg_kernel<<<g, 32, 0, stream>>>(v16s, src, dst, msgpk, alpha, ob);
  };

  // square feature projection (3 -> 10)
  lin_small_kernel<<<(NS * IN0 + 255) / 256, 256, 0, stream>>>(
      square_x, P[104], P[105], sq10, (_Float16*)nullptr, NS, 3, IN0);

  for (int l = 0; l < 3; ++l) {
    int pb = (l == 0) ? 0 : 32 + (l - 1) * 28;
    int vo = (l == 0) ? 28 : 24;

    if (l == 0) {  // K = 10: tiny VALU linears; emit f16 mirrors for k/v
      lin10(piece_x, P[pb + 4],  P[pb + 5],  kd_p, kd_p16, NP);
      lin10(sq10,    P[pb + 6],  P[pb + 7],  kd_s, kd_s16, NS);
      lin10(piece_x, P[pb + 8],  P[pb + 9],  qd_p, nullptr, NP);
      lin10(sq10,    P[pb + 10], P[pb + 11], qd_s, nullptr, NS);
      lin10(piece_x, P[pb + vo + 0], P[pb + vo + 1], vd_p, vd_p16, NP);
      lin10(sq10,    P[pb + vo + 2], P[pb + vo + 3], vd_s, vd_s16, NS);
    } else {
      gemmA16(xp16, P[pb + 4],  P[pb + 5],  nullptr, kd_p, kd_p16, NP, HD, HD, 0, 0);
      gemmA16(xs16, P[pb + 6],  P[pb + 7],  nullptr, kd_s, kd_s16, NS, HD, HD, 0, 0);
      gemmA16(xp16, P[pb + 8],  P[pb + 9],  nullptr, qd_p, nullptr, NP, HD, HD, 0, 0);
      gemmA16(xs16, P[pb + 10], P[pb + 11], nullptr, qd_s, nullptr, NS, HD, HD, 0, 0);
      gemmA16(xp16, P[pb + vo + 0], P[pb + vo + 1], nullptr, vd_p, vd_p16, NP, HD, HD, 0, 0);
      gemmA16(xs16, P[pb + vo + 2], P[pb + vo + 3], nullptr, vd_s, vd_s16, NS, HD, HD, 0, 0);
    }

    fill(out_p, 0.0f, (size_t)NP * HD);
    fill(out_s, 0.0f, (size_t)NS * HD);

    // rel leaves are key-sorted: interacts@12, on@15, ray@18, rev_on@21
    process_edges(ei_int, ei_int + EINT, EINT, kd_p16, qd_p, vd_p16,
                  P[pb + 12], P[pb + 13], P[pb + 14], ea_int, 2, NP, out_p);
    process_edges(ei_ray, ei_ray + ERAY, ERAY, kd_p16, qd_p, vd_p16,
                  P[pb + 18], P[pb + 19], P[pb + 20], nullptr, 0, NP, out_p);
    process_edges(ei_ps, ei_ps + EPS, EPS, kd_p16, qd_s, vd_p16,
                  P[pb + 15], P[pb + 16], P[pb + 17], nullptr, 0, NS, out_s);
    process_edges(ei_sp, ei_sp + ESP, ESP, kd_s16, qd_p, vd_s16,
                  P[pb + 21], P[pb + 22], P[pb + 23], nullptr, 0, NP, out_p);

    const float *skp, *sks;
    if (l == 0) {
      lin10(piece_x, P[pb + 24], P[pb + 25], skip_p, nullptr, NP);
      lin10(sq10,    P[pb + 26], P[pb + 27], skip_s, nullptr, NS);
      skp = skip_p; sks = skip_s;
    } else { skp = xp; sks = xs; }
    // a-linear + skip + GELU; also emits f16 features for the next layer.
    gemmF32(out_p, P[pb + 0], P[pb + 1], skp, xp_new, xp16, NP, HD, HD, 0, 1);
    gemmF32(out_s, P[pb + 2], P[pb + 3], sks, xs_new, xs16, NS, HD, HD, 0, 1);
    float* t;
    t = xp; xp = xp_new; xp_new = t;
    t = xs; xs = xs_new; xs_new = t;
  }

  // ---- ray readout ----
  tval_kernel<<<NP / 4, 128, 0, stream>>>(xp, P[102], P[103], tval);
  gemmA16(xp16, P[100], P[101], nullptr, xproj, nullptr, NP, HD, HD, 0, 0);
  hipMemcpyAsync(xp2, xp, (size_t)NP * HD * sizeof(float),
                 hipMemcpyDeviceToDevice, stream);
  ray_scatter_kernel<<<ERAY, 128, 0, stream>>>(xproj, ea_ray, tval,
                                               ei_ray, ei_ray + ERAY, xp2, ERAY);

  // ---- pooling, GRU, heads ----
  pool_kernel<<<NB, HD, 0, stream>>>(xp2, ppool, NP / NB);
  pool_kernel<<<NB, HD, 0, stream>>>(xs, spool, NS / NB);
  concat_kernel<<<(NB * 2 * HD + 255) / 256, 256, 0, stream>>>(ppool, spool, gbuf);
  // GI = g @ W_ih^T + b_ih   (WMMA, packed-transposed B)
  gemmF32(gbuf, P[92], P[94], nullptr, GI, nullptr, NB, 2 * HD, 3 * HD, 1, 0);
  gru_kernel<<<1, HD, 0, stream>>>(GI, P[93], P[95], gruout);
  heads_kernel<<<NB, HD, 0, stream>>>(gruout,
      P[106], P[107], P[108], P[109],   // win
      P[96],  P[97],  P[98],  P[99],    // mat
      P[88],  P[89],  P[90],  P[91],    // dom
      (float*)d_out);
}
